// DPRNNBaseNew_89489938579878
// MI455X (gfx1250) — matrix operand order repaired and checked by reading, not compiled, not run on hardware
//
#include <hip/hip_runtime.h>

// ============================================================================
// LSTM forward (T=2048, B=64, D=H=256) for MI455X / gfx1250, wave32 + WMMA.
//
// Phase 0 (k_prep):   W_ih,W_hh -> bf16 copies in ws; bias = b_ih+b_hh (f32)
// Phase 1 (k_gatesx): gates_x = x @ W_ih^T + bias, bf16 WMMA, f32 accum,
//                     stored bf16 in *fragment order* [t][tile][lane][8]
// Phase 2 (k_scan):   16 persistent workgroups (one per WGP), each owns 16
//                     h-columns; W_hh slice (32KB bf16) resident in LDS,
//                     c-state in registers, h broadcast via L2 (double
//                     buffered) + one device barrier per timestep.
//
// ws needed: ~258 MB (gates 256MB + weights/bias/hbuf/barrier in first 2MB).
// ============================================================================

#define T_ 2048
#define B_ 64
#define D_ 256
#define H_ 256
#define NWG_ 16
#define SCAN_THREADS 128

// ---- workspace layout (byte offsets) ----
#define BAR_OFF   0u
#define HBUF_OFF  1024u                        // 2 x [64][256] bf16 = 65536 B
#define WIH_OFF   (128u * 1024u)               // 1024*256 bf16 = 524288 B
#define WHH_OFF   (WIH_OFF + 512u * 1024u)
#define BIAS_OFF  (WHH_OFF + 512u * 1024u)     // 1024 f32
#define GATES_OFF (2u * 1024u * 1024u)         // 2048*256*32*16 B = 256 MB

typedef __attribute__((ext_vector_type(16))) __bf16 bf16x16;
typedef __attribute__((ext_vector_type(8)))  float  f32x8;
typedef __attribute__((ext_vector_type(4)))  unsigned u32x4;

union AFrag { bf16x16 v; u32x4 q[2]; };

__device__ __forceinline__ unsigned short f2bf(float f) {
  unsigned u = __float_as_uint(f);
  u += 0x7FFFu + ((u >> 16) & 1u);             // round to nearest even
  return (unsigned short)(u >> 16);
}
__device__ __forceinline__ float fsig(float x) {
  return 1.0f / (1.0f + __expf(-x));
}
__device__ __forceinline__ float ftanh(float x) {
  float e = __expf(-2.0f * fabsf(x));
  float t = (1.0f - e) / (1.0f + e);
  return copysignf(t, x);
}

// ---------------------------------------------------------------------------
// Phase 0: weight conversion + bias sum
// ---------------------------------------------------------------------------
__global__ __launch_bounds__(256) void k_prep(
    const float* __restrict__ wih, const float* __restrict__ whh,
    const float* __restrict__ bih, const float* __restrict__ bhh,
    unsigned short* __restrict__ wih_bf, unsigned short* __restrict__ whh_bf,
    float* __restrict__ bias) {
  int i = blockIdx.x * 256 + threadIdx.x;
  if (i < 4 * H_ * D_) {
    wih_bf[i] = f2bf(wih[i]);
    whh_bf[i] = f2bf(whh[i]);
  }
  if (i < 4 * H_) bias[i] = bih[i] + bhh[i];
}

// ---------------------------------------------------------------------------
// Phase 1: gates_x[t] = x[t] @ W_ih^T + bias   (one workgroup per t)
// 8 waves x 32 tiles (16x16) each, K=256 in 8 k-steps of wmma 16x16x32 bf16.
// Output stored bf16 in fragment order: [t][coltile*4+battile][lane][8].
// ---------------------------------------------------------------------------
__global__ __launch_bounds__(256) void k_gatesx(
    const float* __restrict__ x, const unsigned short* __restrict__ wih_bf,
    const float* __restrict__ bias, unsigned short* __restrict__ gates) {
  __shared__ unsigned short xs[B_ * D_];       // x[t] in bf16, 32 KB

  const int t = blockIdx.x;
  const float* xt = x + (size_t)t * B_ * D_;

  // stage x[t] -> bf16 LDS (row-major [64][256])
  for (int i = threadIdx.x; i < (B_ * D_) / 2; i += 256) {
    float a = xt[2 * i + 0];
    float b = xt[2 * i + 1];
    ((unsigned*)xs)[i] = (unsigned)f2bf(a) | ((unsigned)f2bf(b) << 16);
  }
  __syncthreads();

  const int wave = threadIdx.x >> 5;
  const int lane = threadIdx.x & 31;
  const int lmod = lane & 15;
  const int lhi  = lane >> 4;

  for (int ci = 0; ci < 8; ++ci) {
    const int ct = wave * 8 + ci;              // coltile 0..63
    const int n0 = ct * 16;
    const float bv = bias[n0 + lmod];          // bias depends on N only
    for (int bt = 0; bt < 4; ++bt) {
      const int m0 = bt * 16;
      f32x8 acc = {bv, bv, bv, bv, bv, bv, bv, bv};
#pragma unroll
      for (int k0 = 0; k0 < D_; k0 += 32) {
        // A fragment (x tile, 16x32): lane holds row m0+lmod,
        // K [k0+16*lhi, +16) contiguous
        AFrag A;
        const unsigned short* ap = xs + (m0 + lmod) * D_ + k0 + lhi * 16;
        A.q[0] = *(const u32x4*)ap;
        A.q[1] = *(const u32x4*)(ap + 8);
        // B fragment (W_ih^T tile, 32x16): lane n holds
        // W_ih[n0+lmod][k0+16*lhi .. +16)  (32B contiguous)
        const unsigned short* bp =
            wih_bf + (size_t)(n0 + lmod) * D_ + k0 + lhi * 16;
        bf16x16 Bv = *(const bf16x16*)bp;
        acc = __builtin_amdgcn_wmma_f32_16x16x32_bf16(
            false, A.v, false, Bv, (short)0, acc, false, false);
      }
      // pack accumulator to bf16 fragment, 16B per lane, coalesced per wave
      u32x4 pk;
#pragma unroll
      for (int j = 0; j < 4; ++j)
        pk[j] = (unsigned)f2bf(acc[2 * j]) | ((unsigned)f2bf(acc[2 * j + 1]) << 16);
      const size_t tile = (size_t)(ct * 4 + bt);
      *(u32x4*)(gates + (((size_t)t * 256 + tile) * 32 + lane) * 8) = pk;
    }
  }
}

// ---------------------------------------------------------------------------
// Device-wide barrier across the 16 scan workgroups (monotonic count,
// graph-replay safe: counters zeroed by hipMemsetAsync each launch).
// ---------------------------------------------------------------------------
__device__ __forceinline__ void grid_barrier(unsigned* bar, unsigned step) {
  __threadfence();                             // flush h stores device-scope
  __syncthreads();
  if (threadIdx.x == 0) {
    unsigned prev = __hip_atomic_fetch_add(&bar[0], 1u, __ATOMIC_ACQ_REL,
                                           __HIP_MEMORY_SCOPE_AGENT);
    unsigned target = (unsigned)NWG_ * (step + 1u);
    if (prev + 1u == target) {
      __hip_atomic_store(&bar[1], target, __ATOMIC_RELEASE,
                         __HIP_MEMORY_SCOPE_AGENT);
    } else {
      while (__hip_atomic_load(&bar[1], __ATOMIC_ACQUIRE,
                               __HIP_MEMORY_SCOPE_AGENT) < target)
        __builtin_amdgcn_s_sleep(1);
    }
  }
  __syncthreads();
}

// ---------------------------------------------------------------------------
// Phase 2: sequential LSTM scan. Block g owns h columns [16g, 16g+16).
// 4 waves = 4 batch tiles; each wave holds c (v8f) in registers and computes
// the 4 gate accumulators (i,f,g,o) for its 16x16 h tile -> cell update is
// pure register math. h broadcast through L2 in bf16, double buffered.
// ---------------------------------------------------------------------------
__global__ __launch_bounds__(SCAN_THREADS) void k_scan(
    const unsigned short* __restrict__ gates,
    const unsigned short* __restrict__ whh_bf,
    unsigned short* __restrict__ hbuf,          // [2][64][256] bf16
    float* __restrict__ out, float* __restrict__ hn, float* __restrict__ cn,
    unsigned* __restrict__ bar) {
  __shared__ unsigned short wlds[64 * D_];     // W_hh slice, 32 KB bf16

  const int g    = blockIdx.x;                 // 0..15 : h-column strip
  const int wave = threadIdx.x >> 5;           // 0..3  : batch tile
  const int lane = threadIdx.x & 31;
  const int lmod = lane & 15;
  const int lhi  = lane >> 4;
  const int m0   = wave * 16;
  const int j    = g * 16 + lmod;              // this lane's h column

  // Stage W_hh rows {q*256 + g*16 + jj} -> LDS [q*16+jj][k], once.
  for (int rr = threadIdx.x; rr < 64 * 2; rr += SCAN_THREADS) {
    int row = rr >> 1, half = rr & 1;
    int q = row >> 4, jj = row & 15;
    const unsigned short* src =
        whh_bf + (size_t)(q * H_ + g * 16 + jj) * D_ + half * 128;
    unsigned short* dst = wlds + row * D_ + half * 128;
#pragma unroll
    for (int e = 0; e < 128; e += 8)
      *(u32x4*)(dst + e) = *(const u32x4*)(src + e);
  }
  __syncthreads();

  f32x8 cacc = {0, 0, 0, 0, 0, 0, 0, 0};

  for (int t = 0; t < T_; ++t) {
    const unsigned short* hb = hbuf + (size_t)(t & 1) * (B_ * H_);
    unsigned short* hbn = hbuf + (size_t)((t + 1) & 1) * (B_ * H_);

    // prefetch next timestep's gate fragments into L2
    if (t + 1 < T_) {
#pragma unroll
      for (int q = 0; q < 4; ++q) {
        size_t tile = (size_t)((q * 16 + g) * 4 + wave);
        __builtin_prefetch(
            gates + (((size_t)(t + 1) * 256 + tile) * 32 + lane) * 8, 0, 1);
      }
    }

    f32x8 acc[4];
#pragma unroll
    for (int q = 0; q < 4; ++q) acc[q] = (f32x8){0, 0, 0, 0, 0, 0, 0, 0};

    // G += h @ W_hh^T  for this wave's 16 rows x (4 gates x 16 cols)
#pragma unroll
    for (int k0 = 0; k0 < H_; k0 += 32) {
      AFrag A;  // h tile straight from L2-resident bf16 broadcast buffer
      const unsigned short* ap = hb + (m0 + lmod) * H_ + k0 + lhi * 16;
      A.q[0] = *(const u32x4*)ap;
      A.q[1] = *(const u32x4*)(ap + 8);
#pragma unroll
      for (int q = 0; q < 4; ++q) {
        bf16x16 Bv =
            *(const bf16x16*)(wlds + (q * 16 + lmod) * D_ + k0 + lhi * 16);
        acc[q] = __builtin_amdgcn_wmma_f32_16x16x32_bf16(
            false, A.v, false, Bv, (short)0, acc[q], false, false);
      }
    }

    // G += gates_x fragment (same (vgpr,lane) mapping as phase-1 producer)
#pragma unroll
    for (int q = 0; q < 4; ++q) {
      size_t tile = (size_t)((q * 16 + g) * 4 + wave);
      u32x4 gx =
          *(const u32x4*)(gates + (((size_t)t * 256 + tile) * 32 + lane) * 8);
#pragma unroll
      for (int jj = 0; jj < 4; ++jj) {
        acc[q][2 * jj + 0] += __uint_as_float(gx[jj] << 16);
        acc[q][2 * jj + 1] += __uint_as_float(gx[jj] & 0xFFFF0000u);
      }
    }

    // LSTM cell update entirely in registers
    float* outt = out + (size_t)t * B_ * H_;
#pragma unroll
    for (int e = 0; e < 8; ++e) {
      float iv = fsig(acc[0][e]);
      float fv = fsig(acc[1][e]);
      float gv = ftanh(acc[2][e]);
      float ov = fsig(acc[3][e]);
      float cc = fv * cacc[e] + iv * gv;
      cacc[e] = cc;
      float hh = ov * ftanh(cc);
      int b = m0 + e + 8 * lhi;                // D-matrix layout: M = r (+8)
      outt[(size_t)b * H_ + j] = hh;
      hbn[b * H_ + j] = f2bf(hh);              // broadcast for next step
      if (t == T_ - 1) {
        hn[(size_t)b * H_ + j] = hh;
        cn[(size_t)b * H_ + j] = cc;
      }
    }

    grid_barrier(bar, (unsigned)t);            // publish h[t+1]
  }
}

// ---------------------------------------------------------------------------
extern "C" void kernel_launch(void* const* d_in, const int* in_sizes, int n_in,
                              void* d_out, int out_size, void* d_ws,
                              size_t ws_size, hipStream_t stream) {
  const float* x   = (const float*)d_in[0];
  const float* Wih = (const float*)d_in[1];
  const float* bih = (const float*)d_in[2];
  const float* Whh = (const float*)d_in[3];
  const float* bhh = (const float*)d_in[4];

  char* ws = (char*)d_ws;
  unsigned*       bar   = (unsigned*)(ws + BAR_OFF);
  unsigned short* hbuf  = (unsigned short*)(ws + HBUF_OFF);
  unsigned short* wihb  = (unsigned short*)(ws + WIH_OFF);
  unsigned short* whhb  = (unsigned short*)(ws + WHH_OFF);
  float*          bias  = (float*)(ws + BIAS_OFF);
  unsigned short* gates = (unsigned short*)(ws + GATES_OFF);

  float* out = (float*)d_out;
  float* hn  = out + (size_t)T_ * B_ * H_;
  float* cn  = hn + (size_t)B_ * H_;

  // zero barrier counters + both h buffers (h0 = 0); capture-safe
  hipMemsetAsync(ws, 0, HBUF_OFF + 2u * (B_ * H_ * 2), stream);

  k_prep<<<dim3((4 * H_ * D_) / 256), 256, 0, stream>>>(Wih, Whh, bih, bhh,
                                                        wihb, whhb, bias);
  k_gatesx<<<dim3(T_), 256, 0, stream>>>(x, wihb, bias, gates);
  k_scan<<<dim3(NWG_), SCAN_THREADS, 0, stream>>>(gates, whhb, hbuf, out, hn,
                                                  cn, bar);
}
